// Embedding_8821862826505
// MI455X (gfx1250) — matrix operand order repaired and checked by reading, not compiled, not run on hardware
//
#include <hip/hip_runtime.h>
#include <hip/hip_fp16.h>

#define B_    8
#define N_    2048
#define KNN   20
#define CPAD  352          // padded channel stride (3+64+64+64+128=323 -> 352, mult of 32)
#define EMB   1024
#define EPS_  1e-5f
#define SLOPE 0.2f
#define NEG_INF (-3.0e38f)

// gfx1250 async global->LDS path (guarded: falls back to scalar gather if the
// toolchain doesn't declare the builtins)
#if defined(__has_builtin)
#if __has_builtin(__builtin_amdgcn_global_load_async_to_lds_b128) && \
    __has_builtin(__builtin_amdgcn_s_wait_asynccnt)
#define ASYNC_LDS 1
#endif
#endif

typedef _Float16 h16;
typedef __attribute__((ext_vector_type(16))) _Float16 v16h;
typedef __attribute__((ext_vector_type(8)))  _Float16 v8h;
typedef __attribute__((ext_vector_type(8)))  float    v8f;
typedef __attribute__((ext_vector_type(2)))  float    v2f;
typedef __attribute__((ext_vector_type(4)))  int      v4i;

#ifdef ASYNC_LDS
// builtin signature (from hipcc diagnostic): param0 = v4i* in AS1 (global),
// param1 = LDS destination, params 2/3 = imm offset / cpol
#define GPTR(x) ((__attribute__((address_space(1))) v4i*)(x))
#define LPTR(x) ((__attribute__((address_space(3))) v4i*)(x))
#endif

static __device__ __forceinline__ v16h cat16(v8h a, v8h b) {
  return __builtin_shufflevector(a, b, 0,1,2,3,4,5,6,7,8,9,10,11,12,13,14,15);
}

static __device__ __forceinline__ void atomicMaxF(float* a, float v) {
  // ordered-int trick; destination pre-initialized to -inf
  if (v >= 0.f) atomicMax((int*)a, __float_as_int(v));
  else          atomicMin((unsigned int*)a, __float_as_uint(v));
}

// ---------------- init / utility kernels ----------------

__global__ void k_fill(float* __restrict__ p, float v, int n) {
  for (int i = blockIdx.x * blockDim.x + threadIdx.x; i < n; i += gridDim.x * blockDim.x)
    p[i] = v;
}

// transpose x (B,3,N) into padded feature buffers F (f32) and Fh (f16), zero padding
__global__ void k_init_feat(const float* __restrict__ x, float* __restrict__ F,
                            h16* __restrict__ Fh) {
  int t = blockIdx.x * blockDim.x + threadIdx.x;
  if (t >= B_ * N_) return;
  int b = t / N_, n = t - b * N_;
  float* fr = F  + (size_t)t * CPAD;
  h16*   hr = Fh + (size_t)t * CPAD;
  for (int c = 0; c < CPAD; ++c) {
    float v = (c < 3) ? x[((size_t)b * 3 + c) * N_ + n] : 0.f;
    fr[c] = v; hr[c] = (h16)v;
  }
}

// convert weight (O x I) f32 -> (O x IP) f16 zero-padded
__global__ void k_cvt_w(const float* __restrict__ W, h16* __restrict__ Wh,
                        int I, int IP, int n) {
  int t = blockIdx.x * blockDim.x + threadIdx.x;
  if (t >= n) return;
  int m = t / IP, i = t - m * IP;
  Wh[t] = (i < I) ? (h16)W[(size_t)m * I + i] : (h16)0.f;
}

__global__ void k_sqnorm(const float* __restrict__ F, float* __restrict__ sq, int Cin) {
  int t = blockIdx.x * blockDim.x + threadIdx.x;
  if (t >= B_ * N_) return;
  const float* r = F + (size_t)t * CPAD;
  float s = 0.f;
  for (int c = 0; c < Cin; ++c) s += r[c] * r[c];
  sq[t] = s;
}

__global__ void k_finalize(const float* __restrict__ sums, const float* __restrict__ g,
                           const float* __restrict__ bias, float* __restrict__ scale,
                           float* __restrict__ shift, int O, float invCnt) {
  int c = blockIdx.x * blockDim.x + threadIdx.x;
  if (c >= O) return;
  float mean = sums[c] * invCnt;
  float var  = sums[1024 + c] * invCnt - mean * mean;
  float sc   = g[c] * rsqrtf(var + EPS_);
  scale[c] = sc;
  shift[c] = bias[c] - mean * sc;
}

// ---------------- kNN: f32 WMMA gram -> inv-distance ----------------
// D[i][j] = 2*(xi.xj) - |xi|^2 - |xj|^2 ; self masked to -inf.
// One wave computes a 16x64 tile with V_WMMA_F32_16X16X4_F32.
__global__ __launch_bounds__(128, 1)
void k_dist(const float* __restrict__ Fb, const float* __restrict__ sqb,
            float* __restrict__ D, int Cin) {
  const int tid = threadIdx.x, lane = tid & 31, wave = tid >> 5;
  const int hk = lane >> 4, l15 = lane & 15;
  const int rBase = blockIdx.y * 64 + wave * 16;
  const int cBase = blockIdx.x * 64;
  const v8f vz = {0.f,0.f,0.f,0.f,0.f,0.f,0.f,0.f};
  v8f acc[4] = {vz, vz, vz, vz};
  const int nk = (Cin + 3) >> 2;
  for (int kc = 0; kc < nk; ++kc) {
    // A 16x4 f32: lanes0-15 M=0..15 K={0,1}; lanes16-31 K={2,3}
    const int k0 = kc * 4 + hk * 2;
    v2f a = *(const v2f*)(Fb + (size_t)(rBase + l15) * CPAD + k0);
#pragma unroll
    for (int t = 0; t < 4; ++t) {
      v2f bb = *(const v2f*)(Fb + (size_t)(cBase + t * 16 + l15) * CPAD + k0);
      acc[t] = __builtin_amdgcn_wmma_f32_16x16x4_f32(false, a, false, bb,
                                                     (short)0, acc[t], false, false);
    }
  }
#pragma unroll
  for (int t = 0; t < 4; ++t) {
    const int j = cBase + t * 16 + l15;
    const float sj = sqb[j];
#pragma unroll
    for (int r = 0; r < 8; ++r) {
      const int i = rBase + r + hk * 8;
      float v = 2.f * acc[t][r] - sqb[i] - sj;
      if (i == j) v = NEG_INF;
      D[(size_t)i * N_ + j] = v;
    }
  }
}

// wave-per-row iterative top-20 (argmax + mask) over the LDS-cached row
__global__ __launch_bounds__(128, 1)
void k_topk(const float* __restrict__ D, int* __restrict__ knn) {
  __shared__ float rb[4][N_];
  const int tid = threadIdx.x, lane = tid & 31, wave = tid >> 5;
  const int row = blockIdx.x * 4 + wave;
  float* r = rb[wave];
  for (int t = lane; t < N_; t += 32) r[t] = D[(size_t)row * N_ + t];
  for (int it = 0; it < KNN; ++it) {
    float best = NEG_INF; int bi = 0x7fffffff;
    for (int t = lane; t < N_; t += 32) {
      float v = r[t];
      if (v > best) { best = v; bi = t; }
    }
#pragma unroll
    for (int m = 16; m; m >>= 1) {
      float ov = __shfl_xor(best, m, 32);
      int   oi = __shfl_xor(bi,   m, 32);
      if (ov > best || (ov == best && oi < bi)) { best = ov; bi = oi; }
    }
    if (lane == 0) knn[(size_t)row * KNN + it] = bi;
    if ((bi & 31) == lane) r[bi] = NEG_INF;   // owner lane masks winner
  }
}

// ---------------- edge conv: f16 WMMA GEMM over gathered edge features --------
// Block = 4 waves; covers 8 points (160 cols) x 64 out channels. STATS pass
// accumulates per-channel sum/sumsq; APPLY pass does BN+leakyReLU+max over k.
// Neighbor rows are pulled into LDS with GLOBAL_LOAD_ASYNC_TO_LDS_B128 when the
// toolchain exposes it; the [nbr-ctr ; ctr ; 0] edge feature is then built
// in-place in LDS.
template<bool STATS>
__global__ __launch_bounds__(128, 1)
void k_edgeconv(const h16* __restrict__ Fh, const h16* __restrict__ Wh,
                const int* __restrict__ knn, float* __restrict__ sums,
                const float* __restrict__ scale, const float* __restrict__ shift,
                float* __restrict__ Fout, h16* __restrict__ Fhout,
                int Cin, int IP, int Ooff) {
  __shared__ h16   featLDS[160 * 40];   // 160 cols x 32 chans, stride 40 halves (80B)
  __shared__ h16   ctrLDS[8][224];      // center rows, channels [0,Cin)
  __shared__ float outLDS[64][168];
  const int tid = threadIdx.x, lane = tid & 31, wave = tid >> 5;
  const int hk = lane >> 4, l15 = lane & 15;
  const int b = blockIdx.z, pt = blockIdx.x, mGrp = blockIdx.y;
  const int mBase = mGrp * 64 + wave * 16;
  const int twoC = Cin * 2;
  const v8f vz = {0.f,0.f,0.f,0.f,0.f,0.f,0.f,0.f};
  v8f acc[10];
#pragma unroll
  for (int t = 0; t < 10; ++t) acc[t] = vz;

#ifdef ASYNC_LDS
  // async-preload the 8 center rows once (ASYNCcnt path)
  {
    const int nv8 = (Cin + 7) >> 3;           // 16B packets per row
    for (int i = tid; i < 8 * nv8; i += 128) {
      const int p = i / nv8, v = i - p * nv8;
      const h16* src = Fh + ((size_t)b * N_ + pt * 8 + p) * CPAD + v * 8;
      __builtin_amdgcn_global_load_async_to_lds_b128(GPTR(src), LPTR(&ctrLDS[p][v * 8]),
                                                     0, 0);
    }
  }
#endif

  const int nCh = IP >> 5;
  for (int kc = 0; kc < nCh; ++kc) {
    const int c0 = kc * 32;
#ifdef ASYNC_LDS
    // raw neighbor rows -> featLDS via async DMA (only needed while c0 < Cin)
    if (c0 < Cin) {
      for (int col = tid; col < 160; col += 128) {
        const int p = pt * 8 + col / 20;
        const int kk = col - (col / 20) * 20;
        const int nb = knn[((size_t)b * N_ + p) * KNN + kk];
        const h16* src = Fh + ((size_t)b * N_ + nb) * CPAD + c0;
        h16* dst = featLDS + col * 40;
#pragma unroll
        for (int v = 0; v < 4; ++v)
          __builtin_amdgcn_global_load_async_to_lds_b128(GPTR(src + v * 8),
                                                         LPTR(dst + v * 8), 0, 0);
      }
    }
    __builtin_amdgcn_s_wait_asynccnt(0);
    __syncthreads();
    // in-place transform: raw nbr -> [nbr-ctr ; ctr ; 0]
    for (int col = tid; col < 160; col += 128) {
      const int p = col / 20;
      h16* dst = featLDS + col * 40;
      for (int e = 0; e < 32; ++e) {
        const int ch = c0 + e;
        if (ch < Cin)       dst[e] = (h16)(dst[e] - ctrLDS[p][ch]);
        else if (ch < twoC) dst[e] = ctrLDS[p][ch - Cin];
        else                dst[e] = (h16)0.f;
      }
    }
#else
    // fallback: scalar gather straight from global
    for (int col = tid; col < 160; col += 128) {
      const int p = pt * 8 + col / 20;
      const int kk = col - (col / 20) * 20;
      const int nb = knn[((size_t)b * N_ + p) * KNN + kk];
      const h16* ctr = Fh + ((size_t)b * N_ + p)  * CPAD;
      const h16* nbr = Fh + ((size_t)b * N_ + nb) * CPAD;
      __builtin_prefetch(nbr + c0 + 32, 0, 3);
      h16* dst = featLDS + col * 40;
      for (int e = 0; e < 32; ++e) {
        const int ch = c0 + e;
        h16 v;
        if (ch < Cin)       v = (h16)(nbr[ch] - ctr[ch]);
        else if (ch < twoC) v = ctr[ch - Cin];
        else                v = (h16)0.f;
        dst[e] = v;
      }
    }
#endif
    __syncthreads();
    // A frag: W tile 16x32 f16 (lanes0-15 K base 0, lanes16-31 K base 8; +16 for hi half)
    const h16* wrow = Wh + (size_t)(mBase + l15) * IP + c0 + hk * 8;
    __builtin_prefetch(wrow + 32, 0, 3);
    v8h a0 = *(const v8h*)wrow;
    v8h a1 = *(const v8h*)(wrow + 16);
    v16h A = cat16(a0, a1);
#pragma unroll
    for (int t = 0; t < 10; ++t) {
      const h16* src = featLDS + (t * 16 + l15) * 40 + hk * 8;
      v8h b0 = *(const v8h*)src;
      v8h b1 = *(const v8h*)(src + 16);
      v16h Bf = cat16(b0, b1);
      acc[t] = __builtin_amdgcn_wmma_f32_16x16x32_f16(false, A, false, Bf,
                                                      (short)0, acc[t], false, false);
    }
    __syncthreads();
  }

  if constexpr (STATS) {
#pragma unroll
    for (int r = 0; r < 8; ++r) {
      float s = 0.f, s2 = 0.f;
#pragma unroll
      for (int t = 0; t < 10; ++t) { float v = acc[t][r]; s += v; s2 += v * v; }
#pragma unroll
      for (int m = 1; m < 16; m <<= 1) { s += __shfl_xor(s, m, 32); s2 += __shfl_xor(s2, m, 32); }
      if (l15 == 0) {
        const int ch = mBase + r + hk * 8;
        atomicAdd(&sums[ch], s);
        atomicAdd(&sums[1024 + ch], s2);
      }
    }
  } else {
#pragma unroll
    for (int t = 0; t < 10; ++t)
#pragma unroll
      for (int r = 0; r < 8; ++r) {
        const int ch = mBase + r + hk * 8;
        float v = acc[t][r] * scale[ch] + shift[ch];
        v = v > 0.f ? v : SLOPE * v;
        outLDS[wave * 16 + r + hk * 8][t * 16 + l15] = v;
      }
    __syncthreads();
    for (int o = tid; o < 64 * 8; o += 128) {
      const int lm = o >> 3, lp = o & 7;
      float mx = NEG_INF;
#pragma unroll
      for (int kk = 0; kk < KNN; ++kk) mx = fmaxf(mx, outLDS[lm][lp * 20 + kk]);
      const int ch = mGrp * 64 + lm;
      const int p  = pt * 8 + lp;
      const size_t gi = ((size_t)b * N_ + p) * CPAD + Ooff + ch;
      Fout[gi]  = mx;
      Fhout[gi] = (h16)mx;
    }
  }
}

// ---------------- final 1024x323 conv + BN + leakyReLU + global max ----------
template<bool STATS>
__global__ __launch_bounds__(128, 1)
void k_conv5(const h16* __restrict__ Fh, const h16* __restrict__ W5h,
             float* __restrict__ sums, const float* __restrict__ scale,
             const float* __restrict__ shift, float* __restrict__ out) {
  const int tid = threadIdx.x, lane = tid & 31, wave = tid >> 5;
  const int hk = lane >> 4, l15 = lane & 15;
  const int b = blockIdx.z;
  const int cBase = blockIdx.x * 64;
  const int mBase = blockIdx.y * 64 + wave * 16;
  const v8f vz = {0.f,0.f,0.f,0.f,0.f,0.f,0.f,0.f};
  v8f acc[4] = {vz, vz, vz, vz};
  const int IP = 352;
  for (int kc = 0; kc < 11; ++kc) {
    const h16* wrow = W5h + (size_t)(mBase + l15) * IP + kc * 32 + hk * 8;
    __builtin_prefetch(wrow + 32, 0, 3);
    v8h a0 = *(const v8h*)wrow;
    v8h a1 = *(const v8h*)(wrow + 16);
    v16h A = cat16(a0, a1);
#pragma unroll
    for (int t = 0; t < 4; ++t) {
      const h16* src = Fh + ((size_t)b * N_ + cBase + t * 16 + l15) * CPAD + kc * 32 + hk * 8;
      v8h b0 = *(const v8h*)src;
      v8h b1 = *(const v8h*)(src + 16);
      v16h Bf = cat16(b0, b1);
      acc[t] = __builtin_amdgcn_wmma_f32_16x16x32_f16(false, A, false, Bf,
                                                      (short)0, acc[t], false, false);
    }
  }
  if constexpr (STATS) {
#pragma unroll
    for (int r = 0; r < 8; ++r) {
      float s = 0.f, s2 = 0.f;
#pragma unroll
      for (int t = 0; t < 4; ++t) { float v = acc[t][r]; s += v; s2 += v * v; }
#pragma unroll
      for (int m = 1; m < 16; m <<= 1) { s += __shfl_xor(s, m, 32); s2 += __shfl_xor(s2, m, 32); }
      if (l15 == 0) {
        const int ch = mBase + r + hk * 8;
        atomicAdd(&sums[ch], s);
        atomicAdd(&sums[1024 + ch], s2);
      }
    }
  } else {
#pragma unroll
    for (int r = 0; r < 8; ++r) {
      const int ch = mBase + r + hk * 8;
      float mx = NEG_INF;
#pragma unroll
      for (int t = 0; t < 4; ++t) {
        float v = acc[t][r] * scale[ch] + shift[ch];
        v = v > 0.f ? v : SLOPE * v;
        mx = fmaxf(mx, v);
      }
#pragma unroll
      for (int m = 1; m < 16; m <<= 1) mx = fmaxf(mx, __shfl_xor(mx, m, 32));
      if (l15 == 0) atomicMaxF(&out[(size_t)b * EMB + ch], mx);
    }
  }
}

// ---------------- launch ----------------

extern "C" void kernel_launch(void* const* d_in, const int* in_sizes, int n_in,
                              void* d_out, int out_size, void* d_ws, size_t ws_size,
                              hipStream_t stream) {
  const float* x = (const float*)d_in[0];
  const float* W[5]  = {(const float*)d_in[1], (const float*)d_in[4], (const float*)d_in[7],
                        (const float*)d_in[10], (const float*)d_in[13]};
  const float* g[5]  = {(const float*)d_in[2], (const float*)d_in[5], (const float*)d_in[8],
                        (const float*)d_in[11], (const float*)d_in[14]};
  const float* bb[5] = {(const float*)d_in[3], (const float*)d_in[6], (const float*)d_in[9],
                        (const float*)d_in[12], (const float*)d_in[15]};

  char* p = (char*)d_ws;
  auto take = [&](size_t bytes) -> void* {
    void* r = (void*)p; p += (bytes + 255) & ~size_t(255); return r;
  };
  float* F    = (float*)take((size_t)B_ * N_ * CPAD * 4);
  h16*   Fh   = (h16*)  take((size_t)B_ * N_ * CPAD * 2);
  float* Dst  = (float*)take((size_t)N_ * N_ * 4);          // per-batch slab, reused
  float* sq   = (float*)take((size_t)B_ * N_ * 4);
  int*   knn  = (int*)  take((size_t)B_ * N_ * KNN * 4);
  const int Os[5]  = {64, 64, 64, 128, 1024};
  const int Is[5]  = {6, 134, 262, 390, 323};
  const int IPs[5] = {32, 160, 288, 416, 352};
  h16* Wh[5];
  for (int i = 0; i < 5; ++i) Wh[i] = (h16*)take((size_t)Os[i] * IPs[i] * 2);
  float* stats = (float*)take(2048 * 4);
  float* scale = (float*)take(1024 * 4);
  float* shift = (float*)take(1024 * 4);

  k_init_feat<<<(B_ * N_ + 127) / 128, 128, 0, stream>>>(x, F, Fh);
  for (int i = 0; i < 5; ++i) {
    const int n = Os[i] * IPs[i];
    k_cvt_w<<<(n + 255) / 256, 256, 0, stream>>>(W[i], Wh[i], Is[i], IPs[i], n);
  }

  const int preC[4] = {3, 67, 131, 195};   // input prefix = output channel offset
  const float invEdgeCnt = 1.f / (float)((size_t)B_ * N_ * KNN);
  for (int L = 0; L < 4; ++L) {
    const int Cin = preC[L], O = Os[L], IP = IPs[L];
    k_sqnorm<<<(B_ * N_ + 255) / 256, 256, 0, stream>>>(F, sq, Cin);
    for (int b = 0; b < B_; ++b) {
      k_dist<<<dim3(N_ / 64, N_ / 64), 128, 0, stream>>>(
          F + (size_t)b * N_ * CPAD, sq + (size_t)b * N_, Dst, Cin);
      k_topk<<<N_ / 4, 128, 0, stream>>>(Dst, knn + (size_t)b * N_ * KNN);
    }
    k_fill<<<8, 256, 0, stream>>>(stats, 0.f, 2048);
    dim3 eg(N_ / 8, O / 64, B_);
    k_edgeconv<true><<<eg, 128, 0, stream>>>(Fh, Wh[L], knn, stats, nullptr, nullptr,
                                             nullptr, nullptr, Cin, IP, Cin);
    k_finalize<<<(O + 63) / 64, 64, 0, stream>>>(stats, g[L], bb[L], scale, shift, O,
                                                 invEdgeCnt);
    k_edgeconv<false><<<eg, 128, 0, stream>>>(Fh, Wh[L], knn, stats, scale, shift,
                                              F, Fh, Cin, IP, Cin);
  }

  // layer 5
  k_fill<<<8, 256, 0, stream>>>(stats, 0.f, 2048);
  dim3 cg(N_ / 64, EMB / 64, B_);
  k_conv5<true><<<cg, 128, 0, stream>>>(Fh, Wh[4], stats, nullptr, nullptr, nullptr);
  k_finalize<<<EMB / 64, 64, 0, stream>>>(stats, g[4], bb[4], scale, shift, EMB,
                                          1.f / (float)((size_t)B_ * N_));
  k_fill<<<32, 256, 0, stream>>>((float*)d_out, NEG_INF, out_size);
  k_conv5<false><<<cg, 128, 0, stream>>>(Fh, Wh[4], stats, scale, shift, (float*)d_out);
}
